// Attention_Layer_49838800503251
// MI455X (gfx1250) — matrix operand order repaired and checked by reading, not compiled, not run on hardware
//
#include <hip/hip_runtime.h>
#include <hip/hip_bf16.h>

typedef __attribute__((ext_vector_type(16))) _Float16 v16h;
typedef __attribute__((ext_vector_type(8)))  float    v8f;

#define TT      200
#define DD      64
#define HH1     80
#define HH2     40
#define NTILES  13            // ceil(200/16)
#define TPAD    208           // 13*16
#define NEG_INF (-4294967295.0f)   // -2^32 + 1, matches reference

#define W1S_ELEMS (8*5*32*16)      // kc(8) x ntile(5) x lane(32) x j(16)
#define W2S_ELEMS (3*3*32*16)      // kc2(3) x ntile2(3) x lane(32) x j(16)

// ---------------------------------------------------------------------------
// Pre-kernel: convert W1 [256,80] f32 and W2 [80,40] f32 into fp16,
// pre-swizzled into the per-lane-contiguous WMMA B-fragment layout:
//   element (kc, nt, lane, j)  <-  W[kc*32 + (j&7) + 8*(lane>=16) + 16*(j>=8)][nt*16 + lane%16]
// (B-matrix 32x16 fp16 layout: N = lane%16, K = (j&7) + 8*(lane/16) + 16*(j/8))
// W2 is zero-padded to K=96, N=48.
// ---------------------------------------------------------------------------
__global__ void din_convert_weights(const float* __restrict__ W1,
                                    const float* __restrict__ W2,
                                    _Float16* __restrict__ w1s,
                                    _Float16* __restrict__ w2s) {
    int tid    = blockIdx.x * blockDim.x + threadIdx.x;
    int stride = gridDim.x * blockDim.x;
    for (int e = tid; e < W1S_ELEMS; e += stride) {
        int j    = e & 15;
        int lane = (e >> 4) & 31;
        int rest = e >> 9;
        int nt   = rest % 5;
        int kc   = rest / 5;
        int K    = (j & 7) + ((lane >= 16) ? 8 : 0) + ((j >= 8) ? 16 : 0);
        int r    = kc * 32 + K;            // 0..255
        int c    = nt * 16 + (lane & 15);  // 0..79
        w1s[e] = (_Float16)W1[r * HH1 + c];
    }
    for (int e = tid; e < W2S_ELEMS; e += stride) {
        int j    = e & 15;
        int lane = (e >> 4) & 31;
        int rest = e >> 9;
        int nt   = rest % 3;
        int kc   = rest / 3;
        int K    = (j & 7) + ((lane >= 16) ? 8 : 0) + ((j >= 8) ? 16 : 0);
        int r    = kc * 32 + K;            // 0..95 (pad >=80)
        int c    = nt * 16 + (lane & 15);  // 0..47 (pad >=40)
        float val = (r < HH1 && c < HH2) ? W2[r * HH2 + c] : 0.0f;
        w2s[e] = (_Float16)val;
    }
}

__device__ __forceinline__ float sigmoidf_fast(float x) {
    return 1.0f / (1.0f + __expf(-x));
}

// ---------------------------------------------------------------------------
// Main kernel: one block (4 wave32) per batch element.
// ---------------------------------------------------------------------------
__launch_bounds__(128)
__global__ void din_attention_kernel(const float* __restrict__ q,
                                     const float* __restrict__ k,
                                     const float* __restrict__ v,
                                     const int*   __restrict__ mask,
                                     const float* __restrict__ b1,
                                     const float* __restrict__ b2,
                                     const float* __restrict__ Wf,
                                     const float* __restrict__ bf,
                                     const _Float16* __restrict__ w1s,
                                     const _Float16* __restrict__ w2s,
                                     float* __restrict__ out) {
    __shared__ __align__(32) _Float16 sW1[W1S_ELEMS];       // 40 KB
    __shared__ __align__(32) _Float16 sW2[W2S_ELEMS];       // 9 KB
    __shared__ __align__(16) float    sQ[DD];               // 256 B
    __shared__ __align__(16) float    sK[4][16][DD];        // 16 KB (per-wave k tile)
    __shared__ __align__(32) _Float16 sA2[4][3][32][16];    // 12 KB (per-wave layer-2 A frags)
    __shared__ float sLogit[TPAD];                          // 832 B
    __shared__ float sRed[128];
    __shared__ float sPart[2][DD];

    const int b    = blockIdx.x;
    const int tid  = threadIdx.x;
    const int lane = tid & 31;
    const int wave = tid >> 5;

    // ---- load swizzled fp16 weights into LDS (coalesced 16B chunks) ----
    {
        const uint4* g1 = (const uint4*)w1s;
        uint4*       s1 = (uint4*)sW1;
        for (int i = tid; i < W1S_ELEMS / 8; i += 128) s1[i] = g1[i];
        const uint4* g2 = (const uint4*)w2s;
        uint4*       s2 = (uint4*)sW2;
        for (int i = tid; i < W2S_ELEMS / 8; i += 128) s2[i] = g2[i];
    }
    if (tid < DD) sQ[tid] = q[(size_t)b * DD + tid];
    for (int i = tid; i < TPAD; i += 128) sLogit[i] = NEG_INF;
    __syncthreads();

    // ---- tile loop: wave w handles tiles w, w+4, w+8, w+12 (uniform barriers) ----
    for (int it = 0; it < 4; ++it) {
        const int  tile   = wave + it * 4;
        const bool active = (tile < NTILES);
        const int  t0     = tile * 16;

        // phase 1: stage k[b, t0:t0+16, :] into LDS (float4 coalesced)
        if (active) {
            #pragma unroll
            for (int i = 0; i < 8; ++i) {
                int linear = i * 32 + lane;         // 0..255 float4 slots
                int row    = linear >> 4;           // 16 float4 per row
                int c4     = linear & 15;
                int t      = t0 + row;
                float4 val = make_float4(0.f, 0.f, 0.f, 0.f);
                if (t < TT)
                    val = *(const float4*)(k + ((size_t)b * TT + t) * DD + c4 * 4);
                *(float4*)&sK[wave][row][c4 * 4] = val;
            }
        }
        __syncthreads();

        v8f acc[5];
        // phase 2: layer-1 WMMA (16x256 DIN features x 256x80), sigmoid, stage A2
        if (active) {
            #pragma unroll
            for (int nt = 0; nt < 5; ++nt)
                acc[nt] = (v8f){0.f,0.f,0.f,0.f,0.f,0.f,0.f,0.f};

            const int m   = lane & 15;
            const int hiK = (lane >= 16) ? 8 : 0;

            #pragma unroll
            for (int kc = 0; kc < 8; ++kc) {
                const int seg  = kc >> 1;           // feature segment (q,k,q-k,q*k)
                const int base = (kc & 1) * 32;     // offset within 64-wide segment
                v16h a;
                #pragma unroll
                for (int j = 0; j < 16; ++j) {
                    // A-matrix fp16 layout: K = (j&7) + 8*(lane>=16) + 16*(j>=8)
                    int K   = (j & 7) + hiK + ((j >= 8) ? 16 : 0);
                    int idx = base + K;             // 0..63 within segment
                    float kv = sK[wave][m][idx];
                    float qv = sQ[idx];
                    float val = (seg == 0) ? qv
                              : (seg == 1) ? kv
                              : (seg == 2) ? (qv - kv)
                                           : (qv * kv);
                    a[j] = (_Float16)val;
                }
                #pragma unroll
                for (int nt = 0; nt < 5; ++nt) {
                    v16h bfrag = *(const v16h*)&sW1[((kc * 5 + nt) * 32 + lane) * 16];
                    acc[nt] = __builtin_amdgcn_wmma_f32_16x16x32_f16(
                        false, a, false, bfrag, (short)0, acc[nt], false, false);
                }
            }

            // zero the K=80..95 pad of the layer-2 A staging (chunk 2, j=8..15)
            #pragma unroll
            for (int j = 8; j < 16; ++j) sA2[wave][2][lane][j] = (_Float16)0.f;

            const int mrow = (lane >= 16) ? 8 : 0;
            #pragma unroll
            for (int nt = 0; nt < 5; ++nt) {
                int   col = nt * 16 + (lane & 15);   // hidden unit 0..79
                float b1v = b1[col];
                #pragma unroll
                for (int r = 0; r < 8; ++r) {
                    float h  = sigmoidf_fast(acc[nt][r] + b1v);
                    int   mm = r + mrow;             // row within tile
                    int   kk = col & 31;
                    int   kc2   = col >> 5;
                    int   lane2 = mm + (((kk & 15) >= 8) ? 16 : 0);
                    int   jj    = (kk & 7) + ((kk >= 16) ? 8 : 0);
                    sA2[wave][kc2][lane2][jj] = (_Float16)h;
                }
            }
        }
        __syncthreads();

        // phase 3: layer-2 WMMA (16x96 x 96x48), sigmoid, layer-3 dot, logits
        if (active) {
            v8f acc2[3];
            #pragma unroll
            for (int nt = 0; nt < 3; ++nt)
                acc2[nt] = (v8f){0.f,0.f,0.f,0.f,0.f,0.f,0.f,0.f};

            #pragma unroll
            for (int kc2 = 0; kc2 < 3; ++kc2) {
                v16h a2 = *(const v16h*)&sA2[wave][kc2][lane][0];
                #pragma unroll
                for (int nt2 = 0; nt2 < 3; ++nt2) {
                    v16h b2f = *(const v16h*)&sW2[((kc2 * 3 + nt2) * 32 + lane) * 16];
                    acc2[nt2] = __builtin_amdgcn_wmma_f32_16x16x32_f16(
                        false, a2, false, b2f, (short)0, acc2[nt2], false, false);
                }
            }

            float p[8];
            #pragma unroll
            for (int r = 0; r < 8; ++r) p[r] = 0.f;
            #pragma unroll
            for (int nt2 = 0; nt2 < 3; ++nt2) {
                int   col = nt2 * 16 + (lane & 15);      // 0..47
                float wfv = (col < HH2) ? Wf[col] : 0.0f;
                float b2v = (col < HH2) ? b2[col] : 0.0f;
                #pragma unroll
                for (int r = 0; r < 8; ++r)
                    p[r] += sigmoidf_fast(acc2[nt2][r] + b2v) * wfv;
            }
            // reduce over the 16-lane N group (masks 1,2,4,8 stay within group)
            #pragma unroll
            for (int r = 0; r < 8; ++r) {
                float s = p[r];
                s += __shfl_xor(s, 1);
                s += __shfl_xor(s, 2);
                s += __shfl_xor(s, 4);
                s += __shfl_xor(s, 8);
                p[r] = s;
            }
            if ((lane & 15) == 0) {
                const int mrow = (lane >= 16) ? 8 : 0;
                float bfv = bf[0];
                #pragma unroll
                for (int r = 0; r < 8; ++r) {
                    int t = t0 + r + mrow;
                    if (t < TT) {
                        int mk = mask[(size_t)b * TT + t];
                        sLogit[t] = (mk == 0) ? NEG_INF : (p[r] + bfv);
                    }
                }
            }
        }
        __syncthreads();
    }

    // ---- softmax over T (block-wide) ----
    float lm = NEG_INF;
    for (int i = tid; i < TPAD; i += 128) lm = fmaxf(lm, sLogit[i]);
    sRed[tid] = lm;
    __syncthreads();
    for (int s = 64; s > 0; s >>= 1) {
        if (tid < s) sRed[tid] = fmaxf(sRed[tid], sRed[tid + s]);
        __syncthreads();
    }
    const float mx = sRed[0];
    __syncthreads();

    float ls = 0.f;
    for (int i = tid; i < TPAD; i += 128) {
        float e = __expf(sLogit[i] - mx);   // exp(NEG_INF - mx) -> 0
        sLogit[i] = e;
        ls += e;
    }
    sRed[tid] = ls;
    __syncthreads();
    for (int s = 64; s > 0; s >>= 1) {
        if (tid < s) sRed[tid] += sRed[tid + s];
        __syncthreads();
    }
    const float denom = sRed[0];
    __syncthreads();

    // ---- weighted sum over history: out[b,:] = (attn . v) / denom ----
    {
        const int col  = tid & 63;
        const int half = tid >> 6;
        float accv = 0.f;
        const int tb = half * 100, te = tb + 100;
        for (int t = tb; t < te; ++t)
            accv += sLogit[t] * v[((size_t)b * TT + t) * DD + col];
        sPart[half][col] = accv;
    }
    __syncthreads();
    if (tid < DD)
        out[(size_t)b * DD + tid] = (sPart[0][tid] + sPart[1][tid]) / denom;
}

// ---------------------------------------------------------------------------
// Host launcher
// ---------------------------------------------------------------------------
extern "C" void kernel_launch(void* const* d_in, const int* in_sizes, int n_in,
                              void* d_out, int out_size, void* d_ws, size_t ws_size,
                              hipStream_t stream) {
    const float* q    = (const float*)d_in[0];
    const float* k    = (const float*)d_in[1];
    const float* v    = (const float*)d_in[2];
    const int*   mask = (const int*)d_in[3];
    const float* W1   = (const float*)d_in[4];
    const float* b1   = (const float*)d_in[5];
    const float* W2   = (const float*)d_in[6];
    const float* b2   = (const float*)d_in[7];
    const float* Wf   = (const float*)d_in[8];
    const float* bf   = (const float*)d_in[9];
    float*       out  = (float*)d_out;

    _Float16* w1s = (_Float16*)d_ws;
    _Float16* w2s = w1s + W1S_ELEMS;

    const int B = in_sizes[0] / DD;   // 4096

    din_convert_weights<<<dim3(40), dim3(256), 0, stream>>>(W1, W2, w1s, w2s);
    din_attention_kernel<<<dim3(B), dim3(128), 0, stream>>>(
        q, k, v, mask, b1, b2, Wf, bf, w1s, w2s, out);
}
